// TwistorSelfModifyingRecurrent_19920058319399
// MI455X (gfx1250) — matrix operand order repaired and checked by reading, not compiled, not run on hardware
//
#include <hip/hip_runtime.h>
#include <hip/hip_bf16.h>

// ---------------- problem constants ----------------
#define DI 1024
#define HI 1024
#define BB 4
#define SS 2048
#define BSR (BB * SS)      // 8192 rows of tokens
#define H3 (3 * HI)        // 3072
#define NELEM (BSR * DI)   // 8,388,608

typedef __attribute__((ext_vector_type(16))) __bf16 v16bf;
typedef __attribute__((ext_vector_type(8)))  float  v8f;

union FragU { unsigned u[8]; v16bf v; };

// async global->LDS path (CDNA5), guarded so the file compiles on any toolchain
#if defined(__has_builtin)
#if __has_builtin(__builtin_amdgcn_global_load_async_to_lds_b128) && \
    __has_builtin(__builtin_amdgcn_s_wait_asynccnt)
#define USE_ASYNC_LDS 1
#endif
#endif
#ifndef USE_ASYNC_LDS
#define USE_ASYNC_LDS 0
#endif

// exact parameter type from the compiler diagnostic: vector_size(16) int, addrspace-qualified
typedef int v4i32 __attribute__((vector_size(4 * sizeof(int))));
typedef __attribute__((address_space(1))) v4i32* gv4p;   // global src
typedef __attribute__((address_space(3))) v4i32* lv4p;   // LDS dst

__device__ __forceinline__ unsigned short f2bf(float f) {
  unsigned x = __float_as_uint(f);
  x += 0x7fffu + ((x >> 16) & 1u);        // round-to-nearest-even
  return (unsigned short)(x >> 16);
}
__device__ __forceinline__ float geluf(float x) {
  return 0.5f * x * (1.0f + erff(x * 0.70710678118654752440f));
}
__device__ __forceinline__ float sigm(float x) { return 1.0f / (1.0f + expf(-x)); }

// ---------------- f32 -> bf16 weight conversion ----------------
__global__ __launch_bounds__(256) void k_cvt_bf16(const float* __restrict__ src,
                                                  unsigned short* __restrict__ dst, int n) {
  for (int i = blockIdx.x * 256 + threadIdx.x; i < n; i += gridDim.x * 256)
    dst[i] = f2bf(src[i]);
}

// ---------------- LayerNorm over D=1024, one block per row ----------------
__global__ __launch_bounds__(256) void k_layernorm(const float* __restrict__ x,
                                                   const float* __restrict__ g,
                                                   const float* __restrict__ b,
                                                   float* __restrict__ out) {
  __shared__ float s1[256], s2[256];
  const int m = blockIdx.x, tid = threadIdx.x;
  const float* row = x + (size_t)m * DI;
  float v[4], a = 0.f, sq = 0.f;
#pragma unroll
  for (int j = 0; j < 4; ++j) {
    v[j] = row[tid + j * 256];
    a += v[j]; sq += v[j] * v[j];
  }
  s1[tid] = a; s2[tid] = sq; __syncthreads();
  for (int s = 128; s > 0; s >>= 1) {
    if (tid < s) { s1[tid] += s1[tid + s]; s2[tid] += s2[tid + s]; }
    __syncthreads();
  }
  const float mean = s1[0] * (1.0f / DI);
  const float var  = s2[0] * (1.0f / DI) - mean * mean;
  const float inv  = rsqrtf(var + 1e-5f);
#pragma unroll
  for (int j = 0; j < 4; ++j) {
    int d = tid + j * 256;
    out[(size_t)m * DI + d] = (v[j] - mean) * inv * g[d] + b[d];
  }
}

// ---------------- WMMA bf16 GEMM: C = A(MxK,f32) * W^T (W: NxK bf16) + bias ----------------
// 128x128x32 tile, 256 threads = 8 waves; wave -> 32x64 subtile (2x4 frags, 8 WMMAs/k-step)
// double-buffered LDS; W tile staged with GLOBAL_LOAD_ASYNC_TO_LDS_B128 when available.
#define BMT 128
#define BNT 128
#define BKT 32
#define LDP 40   // padded row length in halves (80B: 16B-aligned rows, conflict-friendly)
enum { EP_BIAS = 0, EP_GELU = 1, EP_RESID = 2 };

template <int EP>
__global__ __launch_bounds__(256) void k_gemm(const float* __restrict__ A,
                                              const unsigned short* __restrict__ W,
                                              const float* __restrict__ bias,
                                              const float* __restrict__ R,
                                              float* __restrict__ C,
                                              int M, int N, int K) {
  __shared__ unsigned short As[2][BMT][LDP];   // 2*128*40*2 = 20 KB
  __shared__ unsigned short Ws[2][BNT][LDP];   // 20 KB
  const int tid = threadIdx.x;
  const int lane = tid & 31;
  const int w = tid >> 5;
  const int n0 = blockIdx.x * BNT, m0 = blockIdx.y * BMT;
  const int wr = (w >> 1) * 32;          // wave row group (0/32/64/96)
  const int wc = (w & 1) * 64;           // wave col group (0/64)
  v8f acc[2][4] = {};

  const int srow = tid >> 1;             // staging row 0..127
  const int scol = (tid & 1) * 16;       // staging col {0,16}

  auto stage = [&](int buf, int k0) {
    // A tile: f32 -> bf16 (16 elements per thread)
    const float* ap = A + (size_t)(m0 + srow) * K + k0 + scol;
    float4 f0 = *(const float4*)ap;
    float4 f1 = *(const float4*)(ap + 4);
    float4 f2 = *(const float4*)(ap + 8);
    float4 f3 = *(const float4*)(ap + 12);
    unsigned short* asd = &As[buf][srow][scol];
    asd[0]  = f2bf(f0.x); asd[1]  = f2bf(f0.y); asd[2]  = f2bf(f0.z); asd[3]  = f2bf(f0.w);
    asd[4]  = f2bf(f1.x); asd[5]  = f2bf(f1.y); asd[6]  = f2bf(f1.z); asd[7]  = f2bf(f1.w);
    asd[8]  = f2bf(f2.x); asd[9]  = f2bf(f2.y); asd[10] = f2bf(f2.z); asd[11] = f2bf(f2.w);
    asd[12] = f2bf(f3.x); asd[13] = f2bf(f3.y); asd[14] = f2bf(f3.z); asd[15] = f2bf(f3.w);
    // W tile: bf16, 32 bytes per thread
    const unsigned short* wp = W + (size_t)(n0 + srow) * K + k0 + scol;
    unsigned short* wsd = &Ws[buf][srow][scol];
#if USE_ASYNC_LDS
    __builtin_amdgcn_global_load_async_to_lds_b128((gv4p)wp,       (lv4p)wsd,       0, 0);
    __builtin_amdgcn_global_load_async_to_lds_b128((gv4p)(wp + 8), (lv4p)(wsd + 8), 0, 0);
#else
    uint4 w0 = *(const uint4*)wp;
    uint4 w1 = *(const uint4*)(wp + 8);
    unsigned* wd = (unsigned*)wsd;
    wd[0] = w0.x; wd[1] = w0.y; wd[2] = w0.z; wd[3] = w0.w;
    wd[4] = w1.x; wd[5] = w1.y; wd[6] = w1.z; wd[7] = w1.w;
#endif
    if (k0 + BKT < K) {
      __builtin_prefetch(wp + BKT, 0, 3);          // near-cache prefetch of next W tile
      __builtin_prefetch(ap + BKT, 0, 3);
    }
  };

  stage(0, 0);
#if USE_ASYNC_LDS
  __builtin_amdgcn_s_wait_asynccnt(0);
#endif
  __syncthreads();

  const int nk = K / BKT;
  const int la = lane & 15;
  const int kab = (lane & 16) ? 16 : 0;   // A frag K-base (contiguous k = 16h + s, matches B)
  const int kbb = (lane & 16) ? 16 : 0;   // B frag K-base (contiguous k = 16h + s, matches A)

  for (int kt = 0; kt < nk; ++kt) {
    const int cur = kt & 1;
    if (kt + 1 < nk) stage(cur ^ 1, (kt + 1) * BKT);

    FragU fa[2], fb[4];
#pragma unroll
    for (int fi = 0; fi < 2; ++fi) {
      const unsigned short* arow = As[cur][wr + fi * 16 + la];
#pragma unroll
      for (int v = 0; v < 8; ++v) {
        int ka = kab + 2 * v;
        fa[fi].u[v] = *(const unsigned*)&arow[ka];
      }
    }
#pragma unroll
    for (int bi = 0; bi < 4; ++bi) {
      const unsigned short* wrow = Ws[cur][wc + bi * 16 + la];
#pragma unroll
      for (int v = 0; v < 8; ++v) {
        fb[bi].u[v] = *(const unsigned*)&wrow[kbb + 2 * v];
      }
    }
#pragma unroll
    for (int fi = 0; fi < 2; ++fi)
#pragma unroll
      for (int bi = 0; bi < 4; ++bi)
        acc[fi][bi] = __builtin_amdgcn_wmma_f32_16x16x32_bf16(
            false, fa[fi].v, false, fb[bi].v, (short)0, acc[fi][bi], false, false);

#if USE_ASYNC_LDS
    __builtin_amdgcn_s_wait_asynccnt(0);
#endif
    __syncthreads();
  }

  // epilogue: f32 C/D layout (lane = N, vgpr r -> M = r + 8*(lane>=16))
  const int mbase = m0 + wr + ((lane & 16) ? 8 : 0);
#pragma unroll
  for (int fi = 0; fi < 2; ++fi) {
#pragma unroll
    for (int bi = 0; bi < 4; ++bi) {
      const int col = n0 + wc + bi * 16 + la;
      const float bb = bias[col];
#pragma unroll
      for (int r = 0; r < 8; ++r) {
        const size_t row = (size_t)(mbase + fi * 16 + r);
        float v = acc[fi][bi][r] + bb;
        if (EP == EP_GELU) v = geluf(v);
        if (EP == EP_RESID) v += R[row * N + col];
        C[row * N + col] = v;
      }
    }
  }
}

// ---------------- persistent GRU scan ----------------
#define GWGS 64

__device__ __forceinline__ void grid_sync(unsigned* cnt, unsigned* gen) {
  __syncthreads();
  if (threadIdx.x == 0) {
    __threadfence();
    unsigned g = __hip_atomic_load(gen, __ATOMIC_RELAXED, __HIP_MEMORY_SCOPE_AGENT);
    if (__hip_atomic_fetch_add(cnt, 1u, __ATOMIC_ACQ_REL, __HIP_MEMORY_SCOPE_AGENT) == GWGS - 1) {
      __hip_atomic_store(cnt, 0u, __ATOMIC_RELAXED, __HIP_MEMORY_SCOPE_AGENT);
      __hip_atomic_store(gen, g + 1u, __ATOMIC_RELEASE, __HIP_MEMORY_SCOPE_AGENT);
    } else {
      while (__hip_atomic_load(gen, __ATOMIC_ACQUIRE, __HIP_MEMORY_SCOPE_AGENT) == g)
        __builtin_amdgcn_s_sleep(2);
    }
  }
  __syncthreads();
}

__global__ __launch_bounds__(256) void k_scan_init(float* hA, float* hB, unsigned* sync) {
  int i = blockIdx.x * 256 + threadIdx.x;
  if (i < BB * HI) { hA[i] = 0.f; hB[i] = 0.f; }
  if (i < 4) sync[i] = 0u;
}

// xi: (B,S,3H) precomputed input GEMM; Whh: (3H,H) bf16; y: (B,S,H)
__global__ __launch_bounds__(256) void k_gru_scan(const float* __restrict__ xi,
                                                  const unsigned short* __restrict__ Whh,
                                                  const float* __restrict__ bhh,
                                                  float* __restrict__ y,
                                                  float* __restrict__ hA,
                                                  float* __restrict__ hB,
                                                  unsigned* __restrict__ sync) {
  __shared__ float hs[BB][HI];         // 16 KB: current hidden state
  __shared__ float gsum[3][BB][16];    // per-gate recurrent sums for our 16 columns
  const int tid = threadIdx.x;
  const int wg = blockIdx.x;           // owns h columns [wg*16, wg*16+16)

  for (int t = 0; t < SS; ++t) {
    const float* hcur = (t & 1) ? hB : hA;
    float* hnxt = (t & 1) ? hA : hB;

    for (int i = tid; i < BB * HI; i += 256)
      hs[i >> 10][i & (HI - 1)] = hcur[i];
    __syncthreads();

    if (tid < 192) {
      const int gate = tid / 64;
      const int rem = tid % 64;
      const int b = rem >> 4;
      const int col = rem & 15;
      const int gcol = wg * 16 + col;
      const unsigned* w2 = (const unsigned*)(Whh + (size_t)(gate * HI + gcol) * HI);
      float sum = 0.f;
#pragma unroll 4
      for (int k2 = 0; k2 < HI / 2; ++k2) {
        unsigned u = w2[k2];
        sum += __uint_as_float(u << 16) * hs[b][2 * k2];
        sum += __uint_as_float(u & 0xffff0000u) * hs[b][2 * k2 + 1];
      }
      gsum[gate][b][col] = sum + bhh[gate * HI + gcol];
    }
    __syncthreads();

    if (tid < 64) {
      const int b = tid >> 4;
      const int col = tid & 15;
      const int gcol = wg * 16 + col;
      const float* xirow = xi + ((size_t)(b * SS + t)) * H3;
      const float r = sigm(xirow[gcol] + gsum[0][b][col]);
      const float z = sigm(xirow[HI + gcol] + gsum[1][b][col]);
      const float n = tanhf(xirow[2 * HI + gcol] + r * gsum[2][b][col]);
      const float hprev = hs[b][gcol];
      const float hnew = (1.f - z) * n + z * hprev;
      hnxt[b * HI + gcol] = hnew;
      y[((size_t)(b * SS + t)) * HI + gcol] = hnew;
    }
    grid_sync(&sync[0], &sync[1]);
  }
}

// ---------------- elementwise / small kernels ----------------
__global__ __launch_bounds__(256) void k_add(float* __restrict__ dst,
                                             const float* __restrict__ a,
                                             const float* __restrict__ b, int n) {
  for (int i = blockIdx.x * 256 + threadIdx.x; i < n; i += gridDim.x * 256)
    dst[i] = a[i] + b[i];
}

// gc[b][d] = mean over S of x[b][s][d]
__global__ __launch_bounds__(256) void k_rowmean(const float* __restrict__ x,
                                                 float* __restrict__ gc) {
  int i = blockIdx.x * 256 + threadIdx.x;   // i over B*D = 4096
  if (i >= BB * DI) return;
  const int b = i / DI, d = i % DI;
  float s = 0.f;
  for (int t = 0; t < SS; ++t) s += x[((size_t)(b * SS + t)) * DI + d];
  gc[i] = s * (1.0f / SS);
}

__global__ __launch_bounds__(256) void k_global_mlp(const float* __restrict__ gc,
                                                    const float* __restrict__ gw1, const float* __restrict__ gb1,
                                                    const float* __restrict__ gw2, const float* __restrict__ gb2,
                                                    const float* __restrict__ gw3, const float* __restrict__ gb3,
                                                    float* __restrict__ gvec) {
  __shared__ float h1[512], h2[256], red[256];
  const int tid = threadIdx.x;
  for (int b = 0; b < BB; ++b) {
    const float* x = gc + b * DI;
    for (int o = tid; o < 512; o += 256) {
      float s = gb1[o];
      const float* w = gw1 + (size_t)o * DI;
      for (int k = 0; k < DI; ++k) s += w[k] * x[k];
      h1[o] = geluf(s);
    }
    __syncthreads();
    {
      float s = gb2[tid];
      const float* w = gw2 + (size_t)tid * 512;
      for (int k = 0; k < 512; ++k) s += w[k] * h1[k];
      h2[tid] = geluf(s);
    }
    __syncthreads();
    red[tid] = gw3[tid] * h2[tid];
    __syncthreads();
    for (int s2 = 128; s2 > 0; s2 >>= 1) {
      if (tid < s2) red[tid] += red[tid + s2];
      __syncthreads();
    }
    if (tid == 0) gvec[b] = sigm(red[0] + gb3[0]);
    __syncthreads();
  }
}

// tf[m] = sigmoid(dot(H[m,:K], w) + bias); one wave (32 lanes) per row
__global__ __launch_bounds__(256) void k_gemv_sig(const float* __restrict__ Hin,
                                                  const float* __restrict__ w,
                                                  const float* __restrict__ bias,
                                                  float* __restrict__ tf, int M, int K) {
  const int wave = threadIdx.x >> 5, lane = threadIdx.x & 31;
  const int m = blockIdx.x * 8 + wave;
  if (m >= M) return;
  const float* row = Hin + (size_t)m * K;
  float s = 0.f;
  for (int k = lane; k < K; k += 32) s += row[k] * w[k];
  for (int off = 16; off; off >>= 1) s += __shfl_down(s, off, 32);
  if (lane == 0) tf[m] = sigm(s + bias[0]);
}

// out[m,:] += coup * mobius_twist(out[m,:]); one block per token row
__global__ __launch_bounds__(256) void k_mobius(float* __restrict__ out,
                                                const float* __restrict__ tf,
                                                const float* __restrict__ gvec,
                                                const float* __restrict__ ar) {
  const int m = blockIdx.x;
  const int b = m / SS;
  float* row = out + (size_t)m * DI;
  const int tid = threadIdx.x;
  const float combined = 0.7f * gvec[b] + 0.3f * tf[m];
  const float coup = 0.1f + ar[0] * (combined - 0.5f) * 2.0f;
  float v[4], p[4];
  const int d0 = tid * 4;
#pragma unroll
  for (int j = 0; j < 4; ++j) {
    const int d = d0 + j;
    v[j] = row[d];
    p[j] = row[(d + DI / 2) & (DI - 1)];   // twist partner: [a,b,c,d] -> [c,-d,-a,b]
  }
  __syncthreads();
#pragma unroll
  for (int j = 0; j < 4; ++j) {
    const int d = d0 + j;
    const float sgn = (d < DI / 4 || d >= 3 * DI / 4) ? 1.0f : -1.0f;
    row[d] = v[j] + coup * sgn * p[j];
  }
}

// ---------------- host-side orchestration ----------------
extern "C" void kernel_launch(void* const* d_in, const int* in_sizes, int n_in,
                              void* d_out, int out_size, void* d_ws, size_t ws_size,
                              hipStream_t stream) {
  (void)in_sizes; (void)n_in; (void)out_size; (void)ws_size;
  const float* x     = (const float*)d_in[0];
  const float* g1w   = (const float*)d_in[1];
  const float* g1b   = (const float*)d_in[2];
  const float* g2w   = (const float*)d_in[3];
  const float* g2b   = (const float*)d_in[4];
  const float* W_ih0 = (const float*)d_in[5];
  const float* W_hh0 = (const float*)d_in[6];
  const float* b_ih0 = (const float*)d_in[7];
  const float* b_hh0 = (const float*)d_in[8];
  const float* W_ih1 = (const float*)d_in[9];
  const float* W_hh1 = (const float*)d_in[10];
  const float* b_ih1 = (const float*)d_in[11];
  const float* b_hh1 = (const float*)d_in[12];
  const float* cw1 = (const float*)d_in[13]; const float* cb1 = (const float*)d_in[14];
  const float* cw2 = (const float*)d_in[15]; const float* cb2 = (const float*)d_in[16];
  const float* cw3 = (const float*)d_in[17]; const float* cb3 = (const float*)d_in[18];
  const float* cw4 = (const float*)d_in[19]; const float* cb4 = (const float*)d_in[20];
  const float* gw1 = (const float*)d_in[21]; const float* gb1 = (const float*)d_in[22];
  const float* gw2 = (const float*)d_in[23]; const float* gb2 = (const float*)d_in[24];
  const float* gw3 = (const float*)d_in[25]; const float* gb3 = (const float*)d_in[26];
  const float* ar  = (const float*)d_in[27];
  const float* fw1 = (const float*)d_in[28]; const float* fb1 = (const float*)d_in[29];
  const float* fw2 = (const float*)d_in[30]; const float* fb2 = (const float*)d_in[31];
  float* out = (float*)d_out;

  // workspace layout
  char* ws = (char*)d_ws;
  size_t off = 0;
  auto take = [&](size_t bytes) -> char* {
    char* p = ws + off;
    off += (bytes + 255) & ~(size_t)255;
    return p;
  };
  unsigned short* wih0b = (unsigned short*)take((size_t)H3 * DI * 2);
  unsigned short* whh0b = (unsigned short*)take((size_t)H3 * HI * 2);
  unsigned short* wih1b = (unsigned short*)take((size_t)H3 * HI * 2);
  unsigned short* whh1b = (unsigned short*)take((size_t)H3 * HI * 2);
  unsigned short* cw1b  = (unsigned short*)take((size_t)DI * DI * 2);
  unsigned short* cw2b  = (unsigned short*)take((size_t)(DI / 2) * DI * 2);
  unsigned short* cw3b  = (unsigned short*)take((size_t)(DI / 4) * (DI / 2) * 2);
  unsigned short* fw1b  = (unsigned short*)take((size_t)(4 * DI) * DI * 2);
  unsigned short* fw2b  = (unsigned short*)take((size_t)DI * (4 * DI) * 2);
  float* bufLN = (float*)take((size_t)NELEM * 4);
  float* x2    = (float*)take((size_t)NELEM * 4);
  float* y0    = (float*)take((size_t)NELEM * 4);
  float* y1    = (float*)take((size_t)NELEM * 4);
  float* big   = (float*)take((size_t)BSR * 4096 * 4);   // xi / coupling chain / FFN hidden
  float* gc    = (float*)take((size_t)BB * DI * 4);
  float* gvec  = (float*)take(256);
  float* hA    = (float*)take((size_t)BB * HI * 4);
  float* hB    = (float*)take((size_t)BB * HI * 4);
  unsigned* syn = (unsigned*)take(256);

  // sub-views of 'big' for the coupling MLP chain
  float* h1c = big;                              // 8192 x 1024
  float* h2c = big + (size_t)BSR * DI;           // 8192 x 512
  float* h3c = h2c + (size_t)BSR * (DI / 2);     // 8192 x 256
  float* tfb = h3c + (size_t)BSR * (DI / 4);     // 8192

  const dim3 blk(256);

  // one-time (per call) weight conversions f32 -> bf16
  k_cvt_bf16<<<512, blk, 0, stream>>>(W_ih0, wih0b, H3 * DI);
  k_cvt_bf16<<<512, blk, 0, stream>>>(W_hh0, whh0b, H3 * HI);
  k_cvt_bf16<<<512, blk, 0, stream>>>(W_ih1, wih1b, H3 * HI);
  k_cvt_bf16<<<512, blk, 0, stream>>>(W_hh1, whh1b, H3 * HI);
  k_cvt_bf16<<<512, blk, 0, stream>>>(cw1, cw1b, DI * DI);
  k_cvt_bf16<<<512, blk, 0, stream>>>(cw2, cw2b, (DI / 2) * DI);
  k_cvt_bf16<<<512, blk, 0, stream>>>(cw3, cw3b, (DI / 4) * (DI / 2));
  k_cvt_bf16<<<512, blk, 0, stream>>>(fw1, fw1b, 4 * DI * DI);
  k_cvt_bf16<<<512, blk, 0, stream>>>(fw2, fw2b, 4 * DI * DI);

  // ---- Block 1: LN -> GRU x2 -> residual ----
  k_layernorm<<<BSR, blk, 0, stream>>>(x, g1w, g1b, bufLN);
  k_gemm<EP_BIAS><<<dim3(H3 / BNT, BSR / BMT), blk, 0, stream>>>(bufLN, wih0b, b_ih0, nullptr, big, BSR, H3, DI);
  k_scan_init<<<16, blk, 0, stream>>>(hA, hB, syn);
  k_gru_scan<<<GWGS, blk, 0, stream>>>(big, whh0b, b_hh0, y0, hA, hB, syn);
  k_gemm<EP_BIAS><<<dim3(H3 / BNT, BSR / BMT), blk, 0, stream>>>(y0, wih1b, b_ih1, nullptr, big, BSR, H3, HI);
  k_scan_init<<<16, blk, 0, stream>>>(hA, hB, syn);
  k_gru_scan<<<GWGS, blk, 0, stream>>>(big, whh1b, b_hh1, y1, hA, hB, syn);
  k_add<<<4096, blk, 0, stream>>>(x2, x, y1, NELEM);

  // ---- Block 2: LN -> adaptive mobius -> residual ----
  k_layernorm<<<BSR, blk, 0, stream>>>(x2, g2w, g2b, bufLN);
  k_rowmean<<<16, blk, 0, stream>>>(bufLN, gc);
  k_global_mlp<<<1, blk, 0, stream>>>(gc, gw1, gb1, gw2, gb2, gw3, gb3, gvec);
  for (int cyc = 0; cyc < 3; ++cyc) {
    k_gemm<EP_GELU><<<dim3(DI / BNT, BSR / BMT), blk, 0, stream>>>(bufLN, cw1b, cb1, nullptr, h1c, BSR, DI, DI);
    k_gemm<EP_GELU><<<dim3((DI / 2) / BNT, BSR / BMT), blk, 0, stream>>>(h1c, cw2b, cb2, nullptr, h2c, BSR, DI / 2, DI);
    k_gemm<EP_GELU><<<dim3((DI / 4) / BNT, BSR / BMT), blk, 0, stream>>>(h2c, cw3b, cb3, nullptr, h3c, BSR, DI / 4, DI / 2);
    k_gemv_sig<<<BSR / 8, blk, 0, stream>>>(h3c, cw4, cb4, tfb, BSR, DI / 4);
    k_mobius<<<BSR, blk, 0, stream>>>(bufLN, tfb, gvec, ar);
  }
  k_add<<<4096, blk, 0, stream>>>(x2, x2, bufLN, NELEM);

  // ---- Block 3: FFN residual ----
  k_gemm<EP_GELU><<<dim3(4096 / BNT, BSR / BMT), blk, 0, stream>>>(x2, fw1b, fb1, nullptr, big, BSR, 4 * DI, DI);
  k_gemm<EP_RESID><<<dim3(DI / BNT, BSR / BMT), blk, 0, stream>>>(big, fw2b, fb2, x2, out, BSR, DI, 4 * DI);
}
